// CNVRegressor_9663676416619
// MI455X (gfx1250) — compile-verified, matrix operand order checked by reading, not run on hardware
//
#include <hip/hip_runtime.h>
#include <hip/hip_bf16.h>
#include <math.h>

typedef __attribute__((ext_vector_type(16))) __bf16 v16bf;
typedef __attribute__((ext_vector_type(8)))  float  v8f;

#define BATCH   16
#define NSEQ    1001     // 1 cls + 1000 positions
#define DMODEL  512
#define HEADS   8
#define DH      64
#define NBF     266
#define FFD     2048
#define MROWS   (BATCH*NSEQ)   // 16016

// ---------------------------------------------------------------- helpers
__device__ inline float gelu_f(float x){
  return 0.5f * x * (1.0f + erff(x * 0.70710678118654752f));
}
__device__ inline unsigned short f2bf(float f){
  unsigned u = __float_as_uint(f);
  unsigned r = u + 0x7FFFu + ((u >> 16) & 1u);   // RNE
  return (unsigned short)(r >> 16);
}
__device__ inline unsigned pack2bf(float a, float b){
  return (unsigned)f2bf(a) | ((unsigned)f2bf(b) << 16);
}
// order-preserving float<->uint for atomic max (handles negatives)
__device__ inline unsigned enc_f(float f){
  unsigned b = __float_as_uint(f);
  return (b & 0x80000000u) ? ~b : (b | 0x80000000u);
}
__device__ inline float dec_f(unsigned e){
  unsigned b = (e & 0x80000000u) ? (e ^ 0x80000000u) : ~e;
  return __uint_as_float(b);
}
union BfPack16 { uint4 u; __bf16 b[8]; };

// ---------------------------------------------------------------- front end
__global__ void mask_kernel(const float* __restrict__ sig, float* __restrict__ mask){
  int idx = blockIdx.x * blockDim.x + threadIdx.x;
  if (idx >= BATCH * NSEQ) return;
  int b = idx / NSEQ, p = idx % NSEQ;
  float m = 1.0f;
  if (p > 0){
    int t = p - 1;
    bool any = false;
    for (int ci = 0; ci < 4; ++ci){
      float s = sig[(b*4 + ci)*1000 + t];
      if (!(s == -1.0f)) any = true;   // NaN != -1 -> true (matches jnp)
    }
    m = any ? 1.0f : 0.0f;
  }
  mask[idx] = m;
}

__global__ void conv1_kernel(const float* __restrict__ sig, const float* __restrict__ w,
                             const float* __restrict__ bias, float* __restrict__ out){
  int idx = blockIdx.x * blockDim.x + threadIdx.x;   // (b,64,1000)
  int t = idx % 1000; int co = (idx / 1000) & 63; int b = idx / 64000;
  float acc = bias[co];
  for (int ci = 0; ci < 4; ++ci){
    const float* sp = sig + (b*4 + ci)*1000;
    const float* wp = w   + (co*4 + ci)*9;
#pragma unroll
    for (int kk = 0; kk < 9; ++kk){
      int tt = t + kk - 4;
      if (tt >= 0 && tt < 1000){
        float s = sp[tt];
        if (!__builtin_isfinite(s) || s == -1.0f) s = 0.0f;
        acc += wp[kk] * s;
      }
    }
  }
  out[idx] = gelu_f(acc);
}

__global__ void conv2_kernel(const float* __restrict__ in, const float* __restrict__ w,
                             const float* __restrict__ bias, float* __restrict__ out){
  int idx = blockIdx.x * blockDim.x + threadIdx.x;   // (b,128,1000)
  int t = idx % 1000; int co = (idx / 1000) & 127; int b = idx / 128000;
  float acc = bias[co];
  for (int ci = 0; ci < 64; ++ci){
    const float* ip = in + (b*64 + ci)*1000;
    const float* wp = w  + (co*64 + ci)*9;
#pragma unroll
    for (int kk = 0; kk < 9; ++kk){
      int tt = t + kk - 4;
      if (tt >= 0 && tt < 1000) acc += wp[kk] * ip[tt];
    }
  }
  out[idx] = gelu_f(acc);
}

__global__ void conv3_kernel(const float* __restrict__ in, const float* __restrict__ w,
                             const float* __restrict__ bias, float* __restrict__ out){
  int idx = blockIdx.x * blockDim.x + threadIdx.x;   // (b,512,1000)
  int t = idx % 1000; int co = (idx / 1000) & 511; int b = idx / 512000;
  float acc = bias[co];
  const float* ip = in + (size_t)b*128*1000 + t;
  const float* wp = w + co*128;
  for (int ci = 0; ci < 128; ++ci) acc += wp[ci] * ip[ci*1000];
  out[idx] = acc;
}

__global__ __launch_bounds__(256)
void gnstats_kernel(const float* __restrict__ x, float* __restrict__ stats){
  int b = blockIdx.x >> 5, g = blockIdx.x & 31;
  const float* p = x + ((size_t)(b*512 + g*16))*1000;   // 16 channels x 1000 contiguous
  float s = 0.f, q = 0.f;
  for (int i = threadIdx.x; i < 16000; i += 256){ float v = p[i]; s += v; q += v*v; }
  __shared__ float rs[256], rq[256];
  rs[threadIdx.x] = s; rq[threadIdx.x] = q; __syncthreads();
  for (int st = 128; st > 0; st >>= 1){
    if (threadIdx.x < st){ rs[threadIdx.x] += rs[threadIdx.x+st]; rq[threadIdx.x] += rq[threadIdx.x+st]; }
    __syncthreads();
  }
  if (threadIdx.x == 0){
    float mu = rs[0] / 16000.f;
    stats[blockIdx.x*2]   = mu;
    stats[blockIdx.x*2+1] = rq[0] / 16000.f - mu*mu;
  }
}

// fused GroupNorm-affine + token LayerNorm + mask + PE + context-embedding + cls
__global__ __launch_bounds__(256)
void seq_build_kernel(const float* __restrict__ x3, const float* __restrict__ gstats,
                      const float* __restrict__ gnw, const float* __restrict__ gnb,
                      const float* __restrict__ lntw, const float* __restrict__ lntb,
                      const float* __restrict__ mask,
                      const float* __restrict__ emb_chr, const float* __restrict__ emb_gene,
                      const float* __restrict__ emb_exon, const float* __restrict__ cls,
                      const int* __restrict__ meta, const int* __restrict__ gene_id,
                      const int* __restrict__ exon_id, float* __restrict__ seq){
  int blk = blockIdx.x;
  int b = blk / NSEQ, p = blk % NSEQ;
  int tid = threadIdx.x;
  int chr = meta[b*3+2]; chr = chr < 0 ? 0 : (chr > 22 ? 22 : chr);
  int gid = gene_id[b];  gid = gid < 0 ? 0 : gid;
  int eid = exon_id[b];  eid = eid < 0 ? 0 : (eid > 128 ? 128 : eid);
  float* out = seq + ((size_t)(b*NSEQ + p))*DMODEL;
  if (p == 0){
    for (int c = tid; c < DMODEL; c += 256){
      float ec = emb_chr[chr*DMODEL+c] + emb_gene[(size_t)gid*DMODEL+c] + emb_exon[eid*DMODEL+c];
      out[c] = cls[c] + ec;
    }
    return;
  }
  int t = p - 1;
  __shared__ float xv[DMODEL];
  __shared__ float rs[256], rq[256];
  float as = 0.f, aq = 0.f;
  for (int c = tid; c < DMODEL; c += 256){
    float raw = x3[((size_t)b*DMODEL + c)*1000 + t];
    int g = c >> 4;
    float mu  = gstats[(b*32 + g)*2];
    float var = gstats[(b*32 + g)*2 + 1];
    float v = (raw - mu) * rsqrtf(var + 1e-5f) * gnw[c] + gnb[c];
    xv[c] = v; as += v; aq += v*v;
  }
  rs[tid] = as; rq[tid] = aq; __syncthreads();
  for (int st = 128; st > 0; st >>= 1){
    if (tid < st){ rs[tid] += rs[tid+st]; rq[tid] += rq[tid+st]; }
    __syncthreads();
  }
  float mu = rs[0] / (float)DMODEL;
  float var = rq[0] / (float)DMODEL - mu*mu;
  float rstd = rsqrtf(var + 1e-5f);
  float mk = mask[b*NSEQ + p];
  for (int c = tid; c < DMODEL; c += 256){
    float tok = (xv[c] - mu) * rstd * lntw[c] + lntb[c];
    tok *= mk;
    float ang = (float)t * expf(-logf(10000.f) * (float)(c & ~1) / (float)DMODEL);
    float pe = (c & 1) ? cosf(ang) : sinf(ang);
    float ec = emb_chr[chr*DMODEL+c] + emb_gene[(size_t)gid*DMODEL+c] + emb_exon[eid*DMODEL+c];
    out[c] = tok + pe + ec;
  }
}

// ---------------------------------------------------------------- layernorm (D=512, 1 block/row)
__global__ __launch_bounds__(256)
void ln_kernel(const float* __restrict__ x, const float* __restrict__ w,
               const float* __restrict__ b, float* __restrict__ y){
  int row = blockIdx.x, tid = threadIdx.x;
  const float* xr = x + (size_t)row * DMODEL;
  float* yr = y + (size_t)row * DMODEL;
  float v0 = xr[tid], v1 = xr[tid+256];
  __shared__ float rs[256], rq[256];
  rs[tid] = v0 + v1; rq[tid] = v0*v0 + v1*v1; __syncthreads();
  for (int st = 128; st > 0; st >>= 1){
    if (tid < st){ rs[tid] += rs[tid+st]; rq[tid] += rq[tid+st]; }
    __syncthreads();
  }
  float mu = rs[0] / (float)DMODEL;
  float var = rq[0] / (float)DMODEL - mu*mu;
  float rstd = rsqrtf(var + 1e-5f);
  yr[tid]     = (v0 - mu) * rstd * w[tid]     + b[tid];
  yr[tid+256] = (v1 - mu) * rstd * w[tid+256] + b[tid+256];
}

// ---------------------------------------------------------------- WMMA bf16 GEMM
// C[M,N] = A[M,K] @ W[K,N] + bias (+ gelu) (+ resid).  Block tile 64x128, BK=32,
// 8 waves; each wave owns a 32x32 patch = 2x2 v_wmma_f32_16x16x32_bf16 tiles.
// LDS: A row-contiguous-K, B TRANSPOSED (col-contiguous-K), both with 96B row
// stride so each fragment is two 16B-aligned ds_load_b128. Epilogue goes via a
// 16x132 LDS slab so global stores are coalesced b128.
#define ALD 48    // ushort stride (96 B, multiple of 16)
#define BLD 48
__global__ __launch_bounds__(256)
void gemm_bf16_wmma(const float* __restrict__ A, const float* __restrict__ W,
                    const float* __restrict__ bias, const float* __restrict__ resid,
                    float* __restrict__ C, int M, int N, int K, int doGelu){
  __shared__ unsigned short As[64][ALD];     // [row][k]
  __shared__ unsigned short Bs[128][BLD];    // [col][k]  (transposed)
  __shared__ float Cs[16][132];
  const int tid  = threadIdx.x;
  const int lane = tid & 31, wid = tid >> 5;
  const int wm = wid >> 2, wn = wid & 3;
  const int bm = blockIdx.y * 64, bn = blockIdx.x * 128;
  const int half = lane >> 4, lrow = lane & 15;

  v8f acc[2][2];
#pragma unroll
  for (int i = 0; i < 2; ++i)
#pragma unroll
    for (int j = 0; j < 2; ++j)
#pragma unroll
      for (int r = 0; r < 8; ++r) acc[i][j][r] = 0.0f;

  const int ar = tid >> 2, ac = (tid & 3) * 8;     // A staging: 8 floats/thread
  const int bkr = tid >> 3, bc0 = (tid & 7) * 16;  // B staging: 16 floats/thread
  const int arow = bm + ar;

  for (int kk = 0; kk < K; kk += 32){
    { // stage A tile 64x32 (fp32 -> bf16), one ds_store_b128 per thread
      float4 a0 = make_float4(0.f,0.f,0.f,0.f), a1 = a0;
      if (arow < M){
        const float4* ap = reinterpret_cast<const float4*>(A + (size_t)arow*K + kk + ac);
        a0 = ap[0]; a1 = ap[1];
      }
      uint4 pk;
      pk.x = pack2bf(a0.x, a0.y); pk.y = pack2bf(a0.z, a0.w);
      pk.z = pack2bf(a1.x, a1.y); pk.w = pack2bf(a1.z, a1.w);
      *reinterpret_cast<uint4*>(&As[ar][ac]) = pk;
    }
    { // stage B tile 32x128 transposed
      const float4* wp4 = reinterpret_cast<const float4*>(W + (size_t)(kk + bkr)*N + bn + bc0);
      float4 b0 = wp4[0], b1 = wp4[1], b2 = wp4[2], b3 = wp4[3];
      float tmp[16] = { b0.x,b0.y,b0.z,b0.w, b1.x,b1.y,b1.z,b1.w,
                        b2.x,b2.y,b2.z,b2.w, b3.x,b3.y,b3.z,b3.w };
#pragma unroll
      for (int j = 0; j < 16; ++j) Bs[bc0 + j][bkr] = f2bf(tmp[j]);
    }
    if (kk + 32 < K){   // speculative prefetch of next tiles (global_prefetch_b8)
      __builtin_prefetch(A + (size_t)arow*K + kk + 32 + ac, 0, 1);
      __builtin_prefetch(W + (size_t)(kk + 32 + bkr)*N + bn + bc0, 0, 1);
    }
    __syncthreads();

    // fragments: two aligned b128 runs each (k = half*8..+7 and half*8+16..+23)
    v16bf af[2], bfm[2];
#pragma unroll
    for (int mi = 0; mi < 2; ++mi){
      int r = wm*32 + mi*16 + lrow;
      BfPack16 lo, hi;
      lo.u = *reinterpret_cast<const uint4*>(&As[r][half*8]);
      hi.u = *reinterpret_cast<const uint4*>(&As[r][half*8 + 16]);
#pragma unroll
      for (int j = 0; j < 8; ++j){ af[mi][j] = lo.b[j]; af[mi][8+j] = hi.b[j]; }
    }
#pragma unroll
    for (int nj = 0; nj < 2; ++nj){
      int cc = wn*32 + nj*16 + lrow;
      BfPack16 lo, hi;
      lo.u = *reinterpret_cast<const uint4*>(&Bs[cc][half*8]);
      hi.u = *reinterpret_cast<const uint4*>(&Bs[cc][half*8 + 16]);
#pragma unroll
      for (int j = 0; j < 8; ++j){ bfm[nj][j] = lo.b[j]; bfm[nj][8+j] = hi.b[j]; }
    }
#pragma unroll
    for (int mi = 0; mi < 2; ++mi)
#pragma unroll
      for (int nj = 0; nj < 2; ++nj)
        acc[mi][nj] = __builtin_amdgcn_wmma_f32_16x16x32_bf16(
            false, af[mi], false, bfm[nj], (short)0, acc[mi][nj], false, false);
    __syncthreads();
  }

  // epilogue: 4 slabs of 16 rows through LDS -> coalesced b128 stores
  for (int s = 0; s < 4; ++s){
    __syncthreads();
    if (wm == (s >> 1)){
      int mi = s & 1;
#pragma unroll
      for (int nj = 0; nj < 2; ++nj)
#pragma unroll
        for (int r = 0; r < 8; ++r)
          Cs[r + half*8][wn*32 + nj*16 + lrow] = acc[mi][nj][r];
    }
    __syncthreads();
    int row = tid >> 4;             // 0..15
    int col = (tid & 15) * 8;       // 0..120
    int m = bm + s*16 + row;
    if (m < M){
      int n0 = bn + col;
      float o[8];
#pragma unroll
      for (int j = 0; j < 8; ++j) o[j] = Cs[row][col + j] + bias[n0 + j];
      if (doGelu){
#pragma unroll
        for (int j = 0; j < 8; ++j) o[j] = gelu_f(o[j]);
      }
      if (resid){
        const float4* rp = reinterpret_cast<const float4*>(resid + (size_t)m*N + n0);
        float4 r0 = rp[0], r1 = rp[1];
        o[0]+=r0.x; o[1]+=r0.y; o[2]+=r0.z; o[3]+=r0.w;
        o[4]+=r1.x; o[5]+=r1.y; o[6]+=r1.z; o[7]+=r1.w;
      }
      float4* cp = reinterpret_cast<float4*>(C + (size_t)m*N + n0);
      cp[0] = make_float4(o[0],o[1],o[2],o[3]);
      cp[1] = make_float4(o[4],o[5],o[6],o[7]);
    }
  }
}

// ---------------------------------------------------------------- FAVOR+
__global__ void kmax_init_kernel(unsigned* p){ *p = 0u; }

__global__ __launch_bounds__(256)
void kdd_max_kernel(const float* __restrict__ k, const float* __restrict__ proj,
                    unsigned* __restrict__ kmax){
  const float dn = 0.35355339059327373f;   // 64^-0.25
  const long long total = (long long)BATCH*HEADS*NSEQ*NBF;
  float lmax = -3.4e38f;
  for (long long idx = (long long)blockIdx.x*256 + threadIdx.x; idx < total;
       idx += (long long)gridDim.x*256){
    int m = (int)(idx % NBF);
    long long r = idx / NBF;
    int n = (int)(r % NSEQ); r /= NSEQ;
    int h = (int)(r % HEADS);
    int b = (int)(r / HEADS);
    const float* kp = k + ((size_t)(b*NSEQ + n))*DMODEL + h*DH;
    const float* pp = proj + m*DH;
    float s = 0.f;
#pragma unroll 8
    for (int d = 0; d < DH; ++d) s += kp[d]*pp[d];
    lmax = fmaxf(lmax, s*dn);
  }
  __shared__ float red[256];
  red[threadIdx.x] = lmax; __syncthreads();
  for (int st = 128; st > 0; st >>= 1){
    if (threadIdx.x < st) red[threadIdx.x] = fmaxf(red[threadIdx.x], red[threadIdx.x+st]);
    __syncthreads();
  }
  if (threadIdx.x == 0) atomicMax(kmax, enc_f(red[0]));
}

// one block per (b,h): ctx[266x64] in registers, ksum in LDS
__global__ __launch_bounds__(256)
void favor_kv_kernel(const float* __restrict__ k, const float* __restrict__ v,
                     const float* __restrict__ mask, const float* __restrict__ proj,
                     const unsigned* __restrict__ kmax_enc,
                     float* __restrict__ g_ctx, float* __restrict__ g_ksum){
  const int b = blockIdx.x >> 3, h = blockIdx.x & 7;
  const int tid = threadIdx.x;
  const float dn = 0.35355339059327373f;
  const float ratio = 0.061313928f;        // 266^-0.5
  __shared__ float krow[DH], vrow[DH], kp_s[NBF], ksum_s[NBF], red[DH];
  float ctxr[67];
#pragma unroll
  for (int i = 0; i < 67; ++i) ctxr[i] = 0.f;
  for (int m = tid; m < NBF; m += 256) ksum_s[m] = 0.f;
  float kstab = dec_f(*kmax_enc);
  __syncthreads();

  for (int n = 0; n < NSEQ; ++n){
    size_t base = ((size_t)(b*NSEQ + n))*DMODEL + h*DH;
    if (tid < DH){
      float kv = k[base + tid];
      krow[tid] = kv; red[tid] = kv*kv;
      vrow[tid] = v[base + tid] * mask[b*NSEQ + n];
    }
    __syncthreads();
    if (tid == 0){ float s = 0.f; for (int i = 0; i < DH; ++i) s += red[i]; red[0] = s; }
    __syncthreads();
    float diag = 0.5f * red[0] * dn * dn;
    for (int m = tid; m < NBF; m += 256){
      float s = 0.f;
      const float* pp = proj + m*DH;
#pragma unroll 8
      for (int d = 0; d < DH; ++d) s += krow[d]*pp[d];
      float kp = ratio * (expf(s*dn - diag - kstab) + 1e-4f);
      kp_s[m] = kp;
      ksum_s[m] += kp;          // same thread owns m each n
    }
    __syncthreads();
#pragma unroll
    for (int i = 0; i < 67; ++i){
      int idx = tid + (i << 8);
      if (idx < NBF*DH) ctxr[i] += kp_s[idx >> 6] * vrow[idx & 63];
    }
    __syncthreads();
  }
  size_t cb = (size_t)blockIdx.x * NBF * DH;
#pragma unroll
  for (int i = 0; i < 67; ++i){
    int idx = tid + (i << 8);
    if (idx < NBF*DH) g_ctx[cb + idx] = ctxr[i];
  }
  for (int m = tid; m < NBF; m += 256) g_ksum[(size_t)blockIdx.x*NBF + m] = ksum_s[m];
}

// one block (64 thr) per (b,h,n): qp row + denominator + out row
__global__ __launch_bounds__(64)
void favor_q_kernel(const float* __restrict__ q, const float* __restrict__ ksum,
                    const float* __restrict__ ctx, const float* __restrict__ proj,
                    float* __restrict__ attn){
  const int blk = blockIdx.x;
  const int n = blk % NSEQ;
  const int bh = blk / NSEQ;             // b*8+h
  const int h = bh & 7, b = bh >> 3;
  const int tid = threadIdx.x;
  const float dn = 0.35355339059327373f;
  const float ratio = 0.061313928f;
  __shared__ float qrow[DH], qp[NBF], red[DH];
  size_t base = ((size_t)(b*NSEQ + n))*DMODEL + h*DH;
  float qv = q[base + tid];
  qrow[tid] = qv; red[tid] = qv*qv;
  __syncthreads();
  if (tid == 0){ float s = 0.f; for (int i = 0; i < DH; ++i) s += red[i]; red[0] = s; }
  __syncthreads();
  float diag = 0.5f * red[0] * dn * dn;
  float lmax = -3.4e38f;
  for (int m = tid; m < NBF; m += 64){
    float s = 0.f;
    const float* pp = proj + m*DH;
#pragma unroll 8
    for (int d = 0; d < DH; ++d) s += qrow[d]*pp[d];
    float dd = s*dn;
    qp[m] = dd;
    lmax = fmaxf(lmax, dd);
  }
  __syncthreads();
  red[tid] = lmax; __syncthreads();
  if (tid == 0){ float s = red[0]; for (int i = 1; i < DH; ++i) s = fmaxf(s, red[i]); red[0] = s; }
  __syncthreads();
  float stab = red[0];
  const float* ks = ksum + (size_t)bh * NBF;
  float dpart = 0.f;
  for (int m = tid; m < NBF; m += 64){
    float e = ratio * (expf(qp[m] - diag - stab) + 1e-4f);
    qp[m] = e;
    dpart += e * ks[m];
  }
  red[tid] = dpart; __syncthreads();
  if (tid == 0){ float s = 0.f; for (int i = 0; i < DH; ++i) s += red[i]; red[0] = s; }
  __syncthreads();
  float dinv = 1.0f / red[0];
  const float* cx = ctx + (size_t)bh * NBF * DH;
  float o = 0.f;
  for (int m = 0; m < NBF; ++m) o += qp[m] * cx[m*DH + tid];
  attn[base + tid] = o * dinv;
}

// ---------------------------------------------------------------- head
__global__ __launch_bounds__(256)
void head_kernel(const float* __restrict__ seq, const float* __restrict__ lw,
                 const float* __restrict__ lb, const float* __restrict__ w1,
                 const float* __restrict__ b1, const float* __restrict__ w2,
                 const float* __restrict__ b2, float* __restrict__ out){
  int b = blockIdx.x, tid = threadIdx.x;
  __shared__ float xl[DMODEL], h1[DMODEL], rs[256], rq[256];
  const float* xr = seq + (size_t)(b*NSEQ) * DMODEL;   // cls row
  float v0 = xr[tid], v1 = xr[tid+256];
  rs[tid] = v0 + v1; rq[tid] = v0*v0 + v1*v1; __syncthreads();
  for (int st = 128; st > 0; st >>= 1){
    if (tid < st){ rs[tid] += rs[tid+st]; rq[tid] += rq[tid+st]; }
    __syncthreads();
  }
  float mu = rs[0] / (float)DMODEL;
  float var = rq[0] / (float)DMODEL - mu*mu;
  float rstd = rsqrtf(var + 1e-5f);
  xl[tid]     = (v0 - mu)*rstd*lw[tid]     + lb[tid];
  xl[tid+256] = (v1 - mu)*rstd*lw[tid+256] + lb[tid+256];
  __syncthreads();
  for (int j = tid; j < DMODEL; j += 256){
    float s = b1[j];
    for (int d = 0; d < DMODEL; ++d) s += xl[d] * w1[(size_t)d*DMODEL + j];
    h1[j] = gelu_f(s);
  }
  __syncthreads();
  float part = 0.f;
  for (int j = tid; j < DMODEL; j += 256) part += h1[j] * w2[j];
  rs[tid] = part; __syncthreads();
  for (int st = 128; st > 0; st >>= 1){
    if (tid < st) rs[tid] += rs[tid+st];
    __syncthreads();
  }
  if (tid == 0) out[b] = rs[0] + b2[0];
}

// ---------------------------------------------------------------- launcher
extern "C" void kernel_launch(void* const* d_in, const int* in_sizes, int n_in,
                              void* d_out, int out_size, void* d_ws, size_t ws_size,
                              hipStream_t stream){
  (void)in_sizes; (void)n_in; (void)out_size; (void)ws_size;
  const float* sig      = (const float*)d_in[0];
  const int*   meta     = (const int*)  d_in[1];
  const int*   gene_id  = (const int*)  d_in[2];
  const int*   exon_id  = (const int*)  d_in[3];
  const float* c1w = (const float*)d_in[4];  const float* c1b = (const float*)d_in[5];
  const float* c2w = (const float*)d_in[6];  const float* c2b = (const float*)d_in[7];
  const float* c3w = (const float*)d_in[8];  const float* c3b = (const float*)d_in[9];
  const float* gnw = (const float*)d_in[10]; const float* gnb = (const float*)d_in[11];
  const float* lntw= (const float*)d_in[12]; const float* lntb= (const float*)d_in[13];
  const float* embc= (const float*)d_in[14]; const float* embg= (const float*)d_in[15];
  const float* embe= (const float*)d_in[16]; const float* cls = (const float*)d_in[17];
  const float* ln1w= (const float*)d_in[18]; const float* ln1b= (const float*)d_in[19];
  const float* wq  = (const float*)d_in[20]; const float* bq  = (const float*)d_in[21];
  const float* wk  = (const float*)d_in[22]; const float* bk  = (const float*)d_in[23];
  const float* wv  = (const float*)d_in[24]; const float* bv  = (const float*)d_in[25];
  const float* wo  = (const float*)d_in[26]; const float* bo  = (const float*)d_in[27];
  const float* proj= (const float*)d_in[28];
  const float* ln2w= (const float*)d_in[29]; const float* ln2b= (const float*)d_in[30];
  const float* fw1 = (const float*)d_in[31]; const float* fb1 = (const float*)d_in[32];
  const float* fw2 = (const float*)d_in[33]; const float* fb2 = (const float*)d_in[34];
  const float* hlw = (const float*)d_in[35]; const float* hlb = (const float*)d_in[36];
  const float* hw1 = (const float*)d_in[37]; const float* hb1 = (const float*)d_in[38];
  const float* hw2 = (const float*)d_in[39]; const float* hb2 = (const float*)d_in[40];
  float* out = (float*)d_out;

  float* Wp = (float*)d_ws;
  size_t o = 0;
  float* mask = Wp + o; o += (size_t)BATCH*NSEQ;
  float* c1   = Wp + o; o += (size_t)BATCH*64*1000;
  float* c2   = Wp + o; o += (size_t)BATCH*128*1000;
  float* x3   = Wp + o; o += (size_t)BATCH*512*1000;
  float* gst  = Wp + o; o += (size_t)BATCH*32*2;
  float* seq  = Wp + o; o += (size_t)MROWS*DMODEL;
  float* lnb  = Wp + o; o += (size_t)MROWS*DMODEL;
  float* qb   = Wp + o; o += (size_t)MROWS*DMODEL;
  float* kb   = Wp + o; o += (size_t)MROWS*DMODEL;
  float* vb   = Wp + o; o += (size_t)MROWS*DMODEL;
  float* ab   = Wp + o; o += (size_t)MROWS*DMODEL;
  float* ffh  = qb;  // alias: q|k|v|attn (4 * M*512 = M*2048 floats), all dead during FF
  float* ctx  = Wp + o; o += (size_t)BATCH*HEADS*NBF*DH;
  float* ksm  = Wp + o; o += (size_t)BATCH*HEADS*NBF;
  unsigned* kmaxu = (unsigned*)(Wp + o); o += 1;

  const int M = MROWS;

  mask_kernel<<<(BATCH*NSEQ + 255)/256, 256, 0, stream>>>(sig, mask);
  conv1_kernel<<<(BATCH*64*1000)/256, 256, 0, stream>>>(sig, c1w, c1b, c1);
  conv2_kernel<<<(BATCH*128*1000)/256, 256, 0, stream>>>(c1, c2w, c2b, c2);
  conv3_kernel<<<(BATCH*512*1000)/256, 256, 0, stream>>>(c2, c3w, c3b, x3);
  gnstats_kernel<<<BATCH*32, 256, 0, stream>>>(x3, gst);
  seq_build_kernel<<<BATCH*NSEQ, 256, 0, stream>>>(x3, gst, gnw, gnb, lntw, lntb, mask,
                                                   embc, embg, embe, cls, meta, gene_id,
                                                   exon_id, seq);

  dim3 g512((512 + 127)/128, (M + 63)/64);     // 4 x 251
  dim3 g2048((FFD + 127)/128, (M + 63)/64);    // 16 x 251

  for (int l = 0; l < 5; ++l){
    const float* Wq = wq + (size_t)l*512*512;  const float* Bq = bq + (size_t)l*512;
    const float* Wk = wk + (size_t)l*512*512;  const float* Bk = bk + (size_t)l*512;
    const float* Wv = wv + (size_t)l*512*512;  const float* Bv = bv + (size_t)l*512;
    const float* Wo = wo + (size_t)l*512*512;  const float* Bo = bo + (size_t)l*512;
    const float* Pj = proj + (size_t)l*NBF*DH;
    const float* F1 = fw1 + (size_t)l*512*FFD; const float* Fb1 = fb1 + (size_t)l*FFD;
    const float* F2 = fw2 + (size_t)l*FFD*512; const float* Fb2 = fb2 + (size_t)l*512;

    ln_kernel<<<M, 256, 0, stream>>>(seq, ln1w + l*512, ln1b + l*512, lnb);
    gemm_bf16_wmma<<<g512, 256, 0, stream>>>(lnb, Wq, Bq, nullptr, qb, M, 512, 512, 0);
    gemm_bf16_wmma<<<g512, 256, 0, stream>>>(lnb, Wk, Bk, nullptr, kb, M, 512, 512, 0);
    gemm_bf16_wmma<<<g512, 256, 0, stream>>>(lnb, Wv, Bv, nullptr, vb, M, 512, 512, 0);

    kmax_init_kernel<<<1, 1, 0, stream>>>(kmaxu);
    kdd_max_kernel<<<2048, 256, 0, stream>>>(kb, Pj, kmaxu);
    favor_kv_kernel<<<BATCH*HEADS, 256, 0, stream>>>(kb, vb, mask, Pj, kmaxu, ctx, ksm);
    favor_q_kernel<<<BATCH*HEADS*NSEQ, 64, 0, stream>>>(qb, ksm, ctx, Pj, ab);

    gemm_bf16_wmma<<<g512, 256, 0, stream>>>(ab, Wo, Bo, seq, seq, M, 512, 512, 0);

    ln_kernel<<<M, 256, 0, stream>>>(seq, ln2w + l*512, ln2b + l*512, lnb);
    gemm_bf16_wmma<<<g2048, 256, 0, stream>>>(lnb, F1, Fb1, nullptr, ffh, M, FFD, 512, 1);
    gemm_bf16_wmma<<<g512, 256, 0, stream>>>(ffh, F2, Fb2, seq, seq, M, 512, FFD, 0);
  }

  head_kernel<<<BATCH, 256, 0, stream>>>(seq, hlw, hlb, hw1, hb1, hw2, hb2, out);
}
